// FuseModel_50276887167011
// MI455X (gfx1250) — matrix operand order repaired and checked by reading, not compiled, not run on hardware
//
#include <hip/hip_runtime.h>

typedef __attribute__((ext_vector_type(16))) _Float16 v16h;
typedef __attribute__((ext_vector_type(8)))  _Float16 v8h;
typedef __attribute__((ext_vector_type(8)))  float    v8f;

namespace {
constexpr int KSZ   = 3;
constexpr int CIN   = 3;
constexpr int HIMG  = 128, WIMG = 128;
constexpr int HO    = 126, WO   = 126;
constexpr int DP    = 27;            // real patch dim = C*K*K
constexpr int DPAD  = 32;            // padded to WMMA K
constexpr int LREAL = HO * WO;       // 15876
constexpr int LPAD  = 16000;         // multiple of 128
constexpr int NTILES = LPAD / 16;    // 1000 query tiles
constexpr int NSPLIT = 4;            // key-dimension splits (occupancy)
constexpr int KSPAN  = LPAD / NSPLIT;  // 4000, multiple of 32
constexpr int WPB   = 8;             // waves per block
constexpr float LOG2E = 1.4426950408889634f;
constexpr float C2    = LOG2E / (0.5f * 0.5f + 0.1f);  // log2 e / 0.35
constexpr float SC2   = 2.0f * C2;
}

// ---------------------------------------------------------------------------
// Kernel 1: extract 3x3 patches from x (keys/values) and y (queries), pack to
// f16 [LPAD][32] (zero-padded), build transposed value matrix [32][LPAD], and
// f32 squared norms. Padded key rows get norm 1e30 -> score -inf -> p = 0.
// ---------------------------------------------------------------------------
__global__ void patch_pack(const float* __restrict__ x, const float* __restrict__ y,
                           _Float16* __restrict__ Xp, _Float16* __restrict__ Rp,
                           _Float16* __restrict__ XpT,
                           float* __restrict__ XX, float* __restrict__ YY) {
  int l = blockIdx.x * blockDim.x + threadIdx.x;
  if (l >= LPAD) return;
  if (l < LREAL) {
    int i = l / WO, j = l % WO;
    float nx = 0.f, ny = 0.f;
#pragma unroll
    for (int d = 0; d < DPAD; ++d) {
      float vx = 0.f, vy = 0.f;
      if (d < DP) {
        int c = d / 9, ki = (d % 9) / 3, kj = d % 3;
        int off = (c * HIMG + (i + ki)) * WIMG + (j + kj);
        vx = x[off]; vy = y[off];
        nx += vx * vx; ny += vy * vy;
      }
      _Float16 hx = (_Float16)vx;
      Xp[l * DPAD + d]  = hx;
      Rp[l * DPAD + d]  = (_Float16)vy;
      XpT[d * LPAD + l] = hx;
    }
    YY[l] = nx;   // |X_m|^2 (column term)
    XX[l] = ny;   // |R_l|^2 (row term)
  } else {
    for (int d = 0; d < DPAD; ++d) {
      Xp[l * DPAD + d] = (_Float16)0.f;
      Rp[l * DPAD + d] = (_Float16)0.f;
      XpT[d * LPAD + l] = (_Float16)0.f;
    }
    YY[l] = 1e30f;  // kill padded keys in softmax
    XX[l] = 1e30f;
  }
}

// 8 f16 at p and 8 f16 at p+16 (16B aligned) -> v16h
__device__ __forceinline__ v16h load_row16(const _Float16* p) {
  v8h lo = *(const v8h*)p;
  v8h hi = *(const v8h*)(p + 16);
  v16h o;
#pragma unroll
  for (int t = 0; t < 8; ++t) { o[t] = lo[t]; o[t + 8] = hi[t]; }
  return o;
}

// WMMA 16-bit A-matrix 16x32 layout: lane<16 holds row (row0+lane), K in
// {0..7, 16..23}; lane>=16 holds row (row0+lane-16), K in {8..15, 24..31}.
__device__ __forceinline__ v16h load_a16(const _Float16* base, int row0, int lane) {
  int r    = row0 + (lane & 15);
  int koff = (lane >> 4) * 8;
  const _Float16* p = base + r * DPAD + koff;
  v8h lo = *(const v8h*)p;
  v8h hi = *(const v8h*)(p + 16);
  v16h o;
#pragma unroll
  for (int t = 0; t < 8; ++t) { o[t] = lo[t]; o[t + 8] = hi[t]; }
  return o;
}

// DPP16 row-local butterfly (lanes 0-15 and 16-31 reduce independently):
// xor1 = quad_perm(1,0,3,2), xor2 = quad_perm(2,3,0,1),
// then row_half_mirror (==xor7) and row_mirror (==xor15) finish the tree.
template <int CTRL>
__device__ __forceinline__ float dpp_mov(float x) {
  return __int_as_float(
      __builtin_amdgcn_mov_dpp(__float_as_int(x), CTRL, 0xF, 0xF, true));
}
__device__ __forceinline__ float half_max16(float x) {
  x = fmaxf(x, dpp_mov<0x0B1>(x));
  x = fmaxf(x, dpp_mov<0x04E>(x));
  x = fmaxf(x, dpp_mov<0x141>(x));
  x = fmaxf(x, dpp_mov<0x140>(x));
  return x;
}
__device__ __forceinline__ float half_sum16(float x) {
  x += dpp_mov<0x0B1>(x);
  x += dpp_mov<0x04E>(x);
  x += dpp_mov<0x141>(x);
  x += dpp_mov<0x140>(x);
  return x;
}

// ---------------------------------------------------------------------------
// Kernel 2: fused flash attention with split-K. One wave per
// (16-row query tile, key range). Per 32-key iteration: 2x score WMMA,
// log2-domain online softmax (DPP reductions), P via LDS, 2x PV WMMA.
// Writes unnormalized partial O plus per-row (max, sum).
// ---------------------------------------------------------------------------
__global__ __launch_bounds__(WPB * 32)
void nl_attention(const _Float16* __restrict__ Xp, const _Float16* __restrict__ Rp,
                  const _Float16* __restrict__ XpT,
                  const float* __restrict__ XX, const float* __restrict__ YY,
                  float* __restrict__ Opart, float* __restrict__ Mpart,
                  float* __restrict__ Spart) {
  __shared__ _Float16 Pstage[WPB][16 * DPAD];   // 1 KB per wave
  const int lane  = threadIdx.x & 31;
  const int wave  = threadIdx.x >> 5;
  const int gw    = blockIdx.x * WPB + wave;    // grid is exact: no guard
  const int tile  = gw >> 2;                    // / NSPLIT
  const int split = gw & (NSPLIT - 1);
  const int l0    = tile * 16;
  const int mBeg  = split * KSPAN;
  const int mEnd  = mBeg + KSPAN;

  const v16h A = load_a16(Rp, l0, lane);        // query tile, loaded once

  // Row-slot scalars follow the C-layout: slot r -> row l0+r (lanes 0-15),
  // row l0+r+8 (lanes 16-31). Pre-scaled into log2 domain.
  float cxx[8];
  const int rbase = l0 + ((lane >= 16) ? 8 : 0);
#pragma unroll
  for (int r = 0; r < 8; ++r) cxx[r] = C2 * XX[rbase + r];

  v8f Oa = {}, Ob = {};                         // output accum: d-cols 0-15 / 16-31
  float rowMax[8], rowSum[8];
#pragma unroll
  for (int r = 0; r < 8; ++r) { rowMax[r] = -3.0e38f; rowSum[r] = 0.0f; }

  _Float16* P = Pstage[wave];
  const int pM0 = (lane >= 16) ? 8 : 0;         // C-layout store coords
  const int pN  = lane & 15;
  const _Float16* pArd = P + (lane & 15) * DPAD + (lane >> 4) * 8;  // A-layout read

  for (int m0 = mBeg; m0 < mEnd; m0 += 32) {
    if (m0 + 32 < mEnd) {
      __builtin_prefetch(Xp + (size_t)(m0 + 32) * DPAD + (size_t)lane * 32, 0, 3);
      __builtin_prefetch(XpT + (size_t)lane * LPAD + (m0 + 32), 0, 3);
    }
    // B operand (key tiles): lane<16 holds col (lane), K {0..7, 16..23};
    // lane>=16 holds col (lane-16), K {8..15, 24..31}. Column n = Xp row (m0+n).
    const _Float16* bp = Xp + (size_t)(m0 + (lane & 15)) * DPAD + (lane >> 4) * 8;
    v16h B0 = load_row16(bp);
    v16h B1 = load_row16(bp + 16 * DPAD);
    v8f zero = {};
    v8f S0 = __builtin_amdgcn_wmma_f32_16x16x32_f16(false, A, false, B0, (short)0, zero, false, false);
    v8f S1 = __builtin_amdgcn_wmma_f32_16x16x32_f16(false, A, false, B1, (short)0, zero, false, false);

    const float cyy0 = C2 * YY[m0 + (lane & 15)];
    const float cyy1 = C2 * YY[m0 + 16 + (lane & 15)];

    // log2-domain scores: t = S*2*C2 - (C2*xx + C2*yy); one FMA per element
    float tmax[8];
#pragma unroll
    for (int r = 0; r < 8; ++r) {
      float b0 = cxx[r] + cyy0;
      float b1 = cxx[r] + cyy1;
      float s0 = __builtin_fmaf(S0[r], SC2, -b0);
      float s1 = __builtin_fmaf(S1[r], SC2, -b1);
      S0[r] = s0; S1[r] = s1;
      tmax[r] = fmaxf(s0, s1);
    }
    // max over the 16 lanes of each half-wave, via DPP (no DS traffic)
#pragma unroll
    for (int r = 0; r < 8; ++r) tmax[r] = half_max16(tmax[r]);

    // online softmax update; rowSum stays lane-distributed (reduced at end)
#pragma unroll
    for (int r = 0; r < 8; ++r) {
      float newM  = fmaxf(rowMax[r], tmax[r]);
      float scale = __builtin_amdgcn_exp2f(rowMax[r] - newM);
      rowMax[r] = newM;
      rowSum[r] *= scale;
      Oa[r] *= scale;
      Ob[r] *= scale;
      float p0 = __builtin_amdgcn_exp2f(S0[r] - newM);
      float p1 = __builtin_amdgcn_exp2f(S1[r] - newM);
      rowSum[r] += p0 + p1;
      S0[r] = p0; S1[r] = p1;
    }
    // stage P (16x32 f16) to LDS in [row][k] order, re-read in A-layout
#pragma unroll
    for (int r = 0; r < 8; ++r) {
      P[(pM0 + r) * DPAD + pN]      = (_Float16)S0[r];
      P[(pM0 + r) * DPAD + pN + 16] = (_Float16)S1[r];
    }
    asm volatile("s_wait_dscnt 0" ::: "memory");
    v16h Pa;
    {
      v8h lo = *(const v8h*)pArd;
      v8h hi = *(const v8h*)(pArd + 16);
#pragma unroll
      for (int t = 0; t < 8; ++t) { Pa[t] = lo[t]; Pa[t + 8] = hi[t]; }
    }
    // V tiles from transposed values: column n (d-index) = XpT row n,
    // K (key index) contiguous along the row.
    const _Float16* vp = XpT + (size_t)(lane & 15) * LPAD + m0 + (lane >> 4) * 8;
    v16h V0 = load_row16(vp);                 // d-cols 0..15
    v16h V1 = load_row16(vp + 16 * LPAD);     // d-cols 16..31
    Oa = __builtin_amdgcn_wmma_f32_16x16x32_f16(false, Pa, false, V0, (short)0, Oa, false, false);
    Ob = __builtin_amdgcn_wmma_f32_16x16x32_f16(false, Pa, false, V1, (short)0, Ob, false, false);
  }

  // finish: reduce rowSum over the half-wave (DPP), write raw partials
#pragma unroll
  for (int r = 0; r < 8; ++r) rowSum[r] = half_sum16(rowSum[r]);

  const int col = lane & 15;
  const size_t rowBase = (size_t)split * LPAD + l0 + pM0;
#pragma unroll
  for (int r = 0; r < 8; ++r) {
    Opart[(rowBase + r) * DPAD + col]      = Oa[r];
    Opart[(rowBase + r) * DPAD + col + 16] = Ob[r];
  }
  if ((lane & 15) == 0) {   // lanes 0 and 16 carry rows l0+0..7 / l0+8..15
#pragma unroll
    for (int r = 0; r < 8; ++r) {
      Mpart[rowBase + r] = rowMax[r];
      Spart[rowBase + r] = rowSum[r];
    }
  }
}

// ---------------------------------------------------------------------------
// Kernel 2b: merge split-K partials: O = sum_s Opart_s*2^(M_s-M) / sum_s S_s*2^(M_s-M)
// ---------------------------------------------------------------------------
__global__ void merge_splits(const float* __restrict__ Opart,
                             const float* __restrict__ Mpart,
                             const float* __restrict__ Spart,
                             float* __restrict__ O) {
  int idx = blockIdx.x * blockDim.x + threadIdx.x;
  if (idx >= LREAL * DPAD) return;
  int l = idx / DPAD;
  int d = idx - l * DPAD;
  float M = -3.0e38f;
#pragma unroll
  for (int s = 0; s < NSPLIT; ++s) M = fmaxf(M, Mpart[(size_t)s * LPAD + l]);
  float S = 0.f, acc = 0.f;
#pragma unroll
  for (int s = 0; s < NSPLIT; ++s) {
    size_t row = (size_t)s * LPAD + l;
    float w = __builtin_amdgcn_exp2f(Mpart[row] - M);
    S   += Spart[row] * w;
    acc += Opart[row * DPAD + d] * w;
  }
  O[(size_t)l * DPAD + d] = acc / S;
}

// ---------------------------------------------------------------------------
// Kernel 3: 3x3 overlap-add with count normalization -> [3][128][128] f32
// ---------------------------------------------------------------------------
__global__ void overlap_add(const float* __restrict__ O, float* __restrict__ out) {
  int idx = blockIdx.x * blockDim.x + threadIdx.x;
  if (idx >= CIN * HIMG * WIMG) return;
  int w = idx % WIMG;
  int h = (idx / WIMG) % HIMG;
  int c = idx / (WIMG * HIMG);
  float acc = 0.f;
  int cnt = 0;
#pragma unroll
  for (int ki = 0; ki < KSZ; ++ki) {
    int i = h - ki;
    if (i < 0 || i >= HO) continue;
#pragma unroll
    for (int kj = 0; kj < KSZ; ++kj) {
      int j = w - kj;
      if (j < 0 || j >= WO) continue;
      int l = i * WO + j;
      acc += O[(size_t)l * DPAD + (c * 9 + ki * 3 + kj)];
      ++cnt;
    }
  }
  out[idx] = acc / (float)cnt;
}

extern "C" void kernel_launch(void* const* d_in, const int* in_sizes, int n_in,
                              void* d_out, int out_size, void* d_ws, size_t ws_size,
                              hipStream_t stream) {
  const float* x = (const float*)d_in[0];
  const float* y = (const float*)d_in[1];
  float* out = (float*)d_out;

  char* ws = (char*)d_ws;
  size_t off = 0;
  auto take = [&](size_t bytes) -> char* {
    char* p = ws + off;
    off += (bytes + 255) & ~size_t(255);
    return p;
  };
  _Float16* Xp  = (_Float16*)take((size_t)LPAD * DPAD * sizeof(_Float16));
  _Float16* Rp  = (_Float16*)take((size_t)LPAD * DPAD * sizeof(_Float16));
  _Float16* XpT = (_Float16*)take((size_t)DPAD * LPAD * sizeof(_Float16));
  float* XX    = (float*)take((size_t)LPAD * sizeof(float));
  float* YY    = (float*)take((size_t)LPAD * sizeof(float));
  float* Opart = (float*)take((size_t)NSPLIT * LPAD * DPAD * sizeof(float));
  float* Mpart = (float*)take((size_t)NSPLIT * LPAD * sizeof(float));
  float* Spart = (float*)take((size_t)NSPLIT * LPAD * sizeof(float));
  float* O     = (float*)take((size_t)LPAD * DPAD * sizeof(float));

  patch_pack<<<(LPAD + 255) / 256, 256, 0, stream>>>(x, y, Xp, Rp, XpT, XX, YY);
  nl_attention<<<(NTILES * NSPLIT) / WPB, WPB * 32, 0, stream>>>(
      Xp, Rp, XpT, XX, YY, Opart, Mpart, Spart);
  merge_splits<<<(LREAL * DPAD + 255) / 256, 256, 0, stream>>>(Opart, Mpart, Spart, O);
  overlap_add<<<(CIN * HIMG * WIMG + 255) / 256, 256, 0, stream>>>(O, out);
}
